// MaskedMGN_35253091565996
// MI455X (gfx1250) — matrix operand order repaired and checked by reading, not compiled, not run on hardware
//
#include <hip/hip_runtime.h>

#define W128 128
#define NN 100000
#define EE 400000
#define DT_C 0.01f
#define EPS_C 1e-5f

typedef __attribute__((ext_vector_type(16))) _Float16 v16h;
typedef __attribute__((ext_vector_type(8)))  _Float16 v8h;
typedef __attribute__((ext_vector_type(8)))  float    v8f;
typedef int vi4 __attribute__((vector_size(16)));

// ---- async global->LDS 16B copy (CDNA5 GLOBAL_LOAD_ASYNC_TO_LDS_B128) -------
static __device__ __forceinline__ void cp16_g2l(void* lds, const void* gbl) {
#if __has_builtin(__builtin_amdgcn_global_load_async_to_lds_b128)
  __builtin_amdgcn_global_load_async_to_lds_b128(
      (__attribute__((address_space(1))) vi4*)(uintptr_t)gbl,
      (__attribute__((address_space(3))) vi4*)(unsigned)(uintptr_t)lds,
      0, 0);
#else
  *(v8h*)lds = *(const v8h*)gbl;
#endif
}

static __device__ __forceinline__ void async_wait() {
#if __has_builtin(__builtin_amdgcn_global_load_async_to_lds_b128)
#if __has_builtin(__builtin_amdgcn_s_wait_asynccnt)
  __builtin_amdgcn_s_wait_asynccnt(0);
#else
  asm volatile("s_wait_asynccnt 0" ::: "memory");
#endif
#endif
}

// ---- WMMA helpers -----------------------------------------------------------
// 16-bit A-matrix 16x32 layout (ISA 7.12.2): lanes 0-15 row M=lane, K 0-7 in
// v[0:3], K 16-23 in v[4:7]; lanes 16-31 row M=lane-16, K 8-15 / K 24-31.
static __device__ __forceinline__ v16h load_a_lds(const _Float16* base, int stride,
                                                  int k0, int lane) {
  int row = lane & 15;
  int kb  = (lane & 16) ? 8 : 0;
  const _Float16* p = base + row * stride + k0 + kb;
  union { v16h v; v8h h[2]; } u;
  u.h[0] = *(const v8h*)p;
  u.h[1] = *(const v8h*)(p + 16);
  return u.v;
}

// 16-bit B-matrix 32x16 layout: lane n<16 holds column n, K 0-7 / K 16-23;
// lanes 16-31 hold column n-16, K 8-15 / K 24-31. wrow = WT + col*K (transposed weight).
static __device__ __forceinline__ v16h load_b_gbl(const _Float16* wrow, int k0, int lane) {
  const _Float16* p = wrow + k0 + ((lane & 16) ? 8 : 0);
  union { v16h v; v8h h[2]; } u;
  u.h[0] = *(const v8h*)p;
  u.h[1] = *(const v8h*)(p + 16);
  return u.v;
}

// One 16x16 output tile (M=16 rows).
template <int K>
static __device__ __forceinline__ v8f wmma_rowtile(const _Float16* aBase, int aStride,
                                                   const _Float16* wT, int lane, int wave) {
  int col = wave * 16 + (lane & 15);
  const _Float16* wrow = wT + (size_t)col * K;
  v8f c = {};
#pragma unroll
  for (int k0 = 0; k0 < K; k0 += 32) {
    v16h a = load_a_lds(aBase, aStride, k0, lane);
    v16h b = load_b_gbl(wrow, k0, lane);
    c = __builtin_amdgcn_wmma_f32_16x16x32_f16(false, a, false, b, (short)0, c,
                                               false, false);
  }
  return c;
}

// Two 16x16 output tiles (M=32 rows), sharing each B fragment across two WMMAs.
template <int K>
static __device__ __forceinline__ void wmma_rowtile2(const _Float16* aBase, int aStride,
                                                     const _Float16* wT, int lane, int wave,
                                                     v8f& cA, v8f& cB) {
  int col = wave * 16 + (lane & 15);
  const _Float16* wrow = wT + (size_t)col * K;
  cA = {};
  cB = {};
#pragma unroll
  for (int k0 = 0; k0 < K; k0 += 32) {
    v16h b   = load_b_gbl(wrow, k0, lane);
    v16h aLo = load_a_lds(aBase, aStride, k0, lane);
    v16h aHi = load_a_lds(aBase + 16 * aStride, aStride, k0, lane);
    cA = __builtin_amdgcn_wmma_f32_16x16x32_f16(false, aLo, false, b, (short)0, cA,
                                                false, false);
    cB = __builtin_amdgcn_wmma_f32_16x16x32_f16(false, aHi, false, b, (short)0, cB,
                                                false, false);
  }
}

// C/D 16x16 f32 layout: VGPR i, lanes 0-15: M=i,N=lane; lanes 16-31: M=i+8.
static __device__ __forceinline__ void store_c_relu_f16(v8f c, _Float16* sH,
                                                        const float* b, int lane, int wave) {
  int n = lane & 15, mh = (lane & 16) ? 8 : 0;
  int col = wave * 16 + n;
  float bias = b[col];
#pragma unroll
  for (int i = 0; i < 8; ++i) {
    float v = c[i] + bias;
    sH[(mh + i) * W128 + col] = (_Float16)(v > 0.f ? v : 0.f);
  }
}

static __device__ __forceinline__ void store_c_f32(v8f c, float* sC,
                                                   const float* b, int lane, int wave) {
  int n = lane & 15, mh = (lane & 16) ? 8 : 0;
  int col = wave * 16 + n;
  float bias = b[col];
#pragma unroll
  for (int i = 0; i < 8; ++i) sC[(mh + i) * W128 + col] = c[i] + bias;
}

// LayerNorm stats over 128 cols, 16 rows, 256 threads.
static __device__ __forceinline__ void ln_stats16(const float* sC, float* sP, float* sP2,
                                                  float* sMu, float* sRs, int tid) {
  int row = tid >> 4, seg = tid & 15;
  const float* p = sC + row * W128 + seg * 8;
  float s = 0.f, s2 = 0.f;
#pragma unroll
  for (int j = 0; j < 8; ++j) { float v = p[j]; s += v; s2 += v * v; }
  sP[tid] = s; sP2[tid] = s2;
  __syncthreads();
  if (tid < 16) {
    float a = 0.f, b = 0.f;
#pragma unroll
    for (int j = 0; j < 16; ++j) { a += sP[tid * 16 + j]; b += sP2[tid * 16 + j]; }
    float mu  = a * (1.f / 128.f);
    float var = b * (1.f / 128.f) - mu * mu;
    sMu[tid] = mu;
    sRs[tid] = rsqrtf(var + EPS_C);
  }
  __syncthreads();
}

// LayerNorm stats over 128 cols, 32 rows, 256 threads.
static __device__ __forceinline__ void ln_stats32(const float* sC, float* sP, float* sP2,
                                                  float* sMu, float* sRs, int tid) {
  int row = tid >> 3, seg = tid & 7;
  const float* p = sC + row * W128 + seg * 16;
  float s = 0.f, s2 = 0.f;
#pragma unroll
  for (int j = 0; j < 16; ++j) { float v = p[j]; s += v; s2 += v * v; }
  sP[tid] = s; sP2[tid] = s2;
  __syncthreads();
  if (tid < 32) {
    float a = 0.f, b = 0.f;
#pragma unroll
    for (int j = 0; j < 8; ++j) { a += sP[tid * 8 + j]; b += sP2[tid * 8 + j]; }
    float mu  = a * (1.f / 128.f);
    float var = b * (1.f / 128.f) - mu * mu;
    sMu[tid] = mu;
    sRs[tid] = rsqrtf(var + EPS_C);
  }
  __syncthreads();
}

static __device__ __forceinline__ void atomic_fadd(float* p, float v) {
  __hip_atomic_fetch_add(p, v, __ATOMIC_RELAXED, __HIP_MEMORY_SCOPE_AGENT);
}

// ---- weight prep: f32 [K][Nc] -> f16 transposed [Nc][K] ---------------------
__global__ void k_transpose_w(const float* __restrict__ w, _Float16* __restrict__ wt,
                              int K, int Nc) {
  int idx = blockIdx.x * 256 + threadIdx.x;
  if (idx >= K * Nc) return;
  int k = idx / Nc, n = idx - k * Nc;
  wt[(size_t)n * K + k] = (_Float16)w[idx];
}

// ---- node encoder: h = LN(relu(x@W1+b1)@W2+b2)*g+be (M=16) ------------------
__global__ void __launch_bounds__(256) k_node_encoder(
    const float* __restrict__ x, const float* __restrict__ W1, const float* __restrict__ b1,
    const _Float16* __restrict__ W2T, const float* __restrict__ b2,
    const float* __restrict__ g, const float* __restrict__ be,
    float* __restrict__ h, _Float16* __restrict__ h16) {
  __shared__ float sX[16 * 3];
  __shared__ __attribute__((aligned(16))) _Float16 sH[16 * W128];
  __shared__ float sC[16 * W128];
  __shared__ float sP[256], sP2[256], sMu[16], sRs[16];
  int tid = threadIdx.x, lane = tid & 31, wave = tid >> 5;
  int tile = blockIdx.x * 16;
  if (tid < 48) sX[tid] = x[(size_t)tile * 3 + tid];
  __syncthreads();
  {
    int row = tid >> 4, c0 = (tid & 15) * 8;
    float x0 = sX[row * 3], x1 = sX[row * 3 + 1], x2 = sX[row * 3 + 2];
#pragma unroll
    for (int j = 0; j < 8; ++j) {
      int c = c0 + j;
      float v = b1[c] + x0 * W1[c] + x1 * W1[W128 + c] + x2 * W1[2 * W128 + c];
      sH[row * W128 + c] = (_Float16)(v > 0.f ? v : 0.f);
    }
  }
  __syncthreads();
  v8f c = wmma_rowtile<128>(sH, W128, W2T, lane, wave);
  store_c_f32(c, sC, b2, lane, wave);
  __syncthreads();
  ln_stats16(sC, sP, sP2, sMu, sRs, tid);
  {
    int row = tid >> 4, c0 = (tid & 15) * 8;
    float mu = sMu[row], rs = sRs[row];
    size_t base = (size_t)(tile + row) * W128;
#pragma unroll
    for (int j = 0; j < 8; ++j) {
      int cc = c0 + j;
      float v = (sC[row * W128 + cc] - mu) * rs * g[cc] + be[cc];
      h[base + cc] = v;
      h16[base + cc] = (_Float16)v;
    }
  }
}

// ---- edge encoder: e = LN(relu(ea@W1+b1)@W2+b2)*g+be (f16 out, M=16) --------
__global__ void __launch_bounds__(256) k_edge_encoder(
    const float* __restrict__ ea, const float* __restrict__ W1, const float* __restrict__ b1,
    const _Float16* __restrict__ W2T, const float* __restrict__ b2,
    const float* __restrict__ g, const float* __restrict__ be,
    _Float16* __restrict__ e16) {
  __shared__ float sX[16 * 2];
  __shared__ __attribute__((aligned(16))) _Float16 sH[16 * W128];
  __shared__ float sC[16 * W128];
  __shared__ float sP[256], sP2[256], sMu[16], sRs[16];
  int tid = threadIdx.x, lane = tid & 31, wave = tid >> 5;
  int tile = blockIdx.x * 16;
  if (tid < 32) sX[tid] = ea[(size_t)tile * 2 + tid];
  __syncthreads();
  {
    int row = tid >> 4, c0 = (tid & 15) * 8;
    float x0 = sX[row * 2], x1 = sX[row * 2 + 1];
#pragma unroll
    for (int j = 0; j < 8; ++j) {
      int c = c0 + j;
      float v = b1[c] + x0 * W1[c] + x1 * W1[W128 + c];
      sH[row * W128 + c] = (_Float16)(v > 0.f ? v : 0.f);
    }
  }
  __syncthreads();
  v8f c = wmma_rowtile<128>(sH, W128, W2T, lane, wave);
  store_c_f32(c, sC, b2, lane, wave);
  __syncthreads();
  ln_stats16(sC, sP, sP2, sMu, sRs, tid);
  {
    int row = tid >> 4, c0 = (tid & 15) * 8;
    float mu = sMu[row], rs = sRs[row];
    size_t base = (size_t)(tile + row) * W128;
#pragma unroll
    for (int j = 0; j < 8; ++j) {
      int cc = c0 + j;
      float v = (sC[row * W128 + cc] - mu) * rs * g[cc] + be[cc];
      e16[base + cc] = (_Float16)v;
    }
  }
}

// ---- edge layer (M=32): e += MLP_LN([e|h(src)|h(dst)]); agg[dst] += e -------
__global__ void __launch_bounds__(256) k_edge_layer(
    const int* __restrict__ eidx, _Float16* __restrict__ e16,
    const _Float16* __restrict__ h16, float* __restrict__ agg,
    const _Float16* __restrict__ W1T, const float* __restrict__ b1,
    const _Float16* __restrict__ W2T, const float* __restrict__ b2,
    const float* __restrict__ g, const float* __restrict__ be) {
  __shared__ int sSrc[32], sDst[32];
  __shared__ __attribute__((aligned(16))) _Float16 sA[32 * 384];   // 24KB
  __shared__ __attribute__((aligned(16))) _Float16 sH[32 * W128];  // 8KB
  __shared__ float sC[32 * W128];                                  // 16KB
  __shared__ float sP[256], sP2[256], sMu[32], sRs[32];
  int tid = threadIdx.x, lane = tid & 31, wave = tid >> 5;
  int tile = blockIdx.x * 32;
  if (tid < 32) { sSrc[tid] = eidx[tile + tid]; sDst[tid] = eidx[EE + tile + tid]; }
  __syncthreads();
  {
    int r0 = tid >> 4, ch = tid & 15;
#pragma unroll
    for (int r2 = 0; r2 < 32; r2 += 16) {
      int row = r0 + r2;
      cp16_g2l(&sA[row * 384 + ch * 8],
               &e16[(size_t)(tile + row) * W128 + ch * 8]);
      cp16_g2l(&sA[row * 384 + 128 + ch * 8],
               &h16[(size_t)sSrc[row] * W128 + ch * 8]);
      cp16_g2l(&sA[row * 384 + 256 + ch * 8],
               &h16[(size_t)sDst[row] * W128 + ch * 8]);
    }
  }
  async_wait();
  __syncthreads();
  v8f c1a, c1b;
  wmma_rowtile2<384>(sA, 384, W1T, lane, wave, c1a, c1b);
  store_c_relu_f16(c1a, sH, b1, lane, wave);
  store_c_relu_f16(c1b, sH + 16 * W128, b1, lane, wave);
  __syncthreads();
  v8f c2a, c2b;
  wmma_rowtile2<128>(sH, W128, W2T, lane, wave, c2a, c2b);
  store_c_f32(c2a, sC, b2, lane, wave);
  store_c_f32(c2b, sC + 16 * W128, b2, lane, wave);
  __syncthreads();
  ln_stats32(sC, sP, sP2, sMu, sRs, tid);
  {
    int row = tid >> 3, c0 = (tid & 7) * 16;
    float mu = sMu[row], rs = sRs[row];
    int d = sDst[row];
    size_t ebase = (size_t)(tile + row) * W128;
    size_t abase = (size_t)d * W128;
#pragma unroll
    for (int j = 0; j < 16; ++j) {
      int cc = c0 + j;
      float v = (sC[row * W128 + cc] - mu) * rs * g[cc] + be[cc];
      float en = (float)e16[ebase + cc] + v;
      e16[ebase + cc] = (_Float16)en;
      atomic_fadd(&agg[abase + cc], en);
    }
  }
}

// ---- node layer (M=32): h += MLP_LN([h|agg]) --------------------------------
__global__ void __launch_bounds__(256) k_node_layer(
    float* __restrict__ h, _Float16* __restrict__ h16, const float* __restrict__ agg,
    const _Float16* __restrict__ W1T, const float* __restrict__ b1,
    const _Float16* __restrict__ W2T, const float* __restrict__ b2,
    const float* __restrict__ g, const float* __restrict__ be) {
  __shared__ __attribute__((aligned(16))) _Float16 sA[32 * 256];   // 16KB
  __shared__ __attribute__((aligned(16))) _Float16 sH[32 * W128];  // 8KB
  __shared__ float sC[32 * W128];                                  // 16KB
  __shared__ float sP[256], sP2[256], sMu[32], sRs[32];
  int tid = threadIdx.x, lane = tid & 31, wave = tid >> 5;
  int tile = blockIdx.x * 32;
  {
    int r0 = tid >> 4, ch = tid & 15;
#pragma unroll
    for (int r2 = 0; r2 < 32; r2 += 16) {
      int row = r0 + r2;
      cp16_g2l(&sA[row * 256 + ch * 8],
               &h16[(size_t)(tile + row) * W128 + ch * 8]);
      const float* ap = &agg[(size_t)(tile + row) * W128 + ch * 8];
      _Float16* op = &sA[row * 256 + 128 + ch * 8];
#pragma unroll
      for (int j = 0; j < 8; ++j) op[j] = (_Float16)ap[j];
    }
  }
  async_wait();
  __syncthreads();
  v8f c1a, c1b;
  wmma_rowtile2<256>(sA, 256, W1T, lane, wave, c1a, c1b);
  store_c_relu_f16(c1a, sH, b1, lane, wave);
  store_c_relu_f16(c1b, sH + 16 * W128, b1, lane, wave);
  __syncthreads();
  v8f c2a, c2b;
  wmma_rowtile2<128>(sH, W128, W2T, lane, wave, c2a, c2b);
  store_c_f32(c2a, sC, b2, lane, wave);
  store_c_f32(c2b, sC + 16 * W128, b2, lane, wave);
  __syncthreads();
  ln_stats32(sC, sP, sP2, sMu, sRs, tid);
  {
    int row = tid >> 3, c0 = (tid & 7) * 16;
    float mu = sMu[row], rs = sRs[row];
    size_t base = (size_t)(tile + row) * W128;
#pragma unroll
    for (int j = 0; j < 16; ++j) {
      int cc = c0 + j;
      float v = (sC[row * W128 + cc] - mu) * rs * g[cc] + be[cc];
      float hn = h[base + cc] + v;
      h[base + cc] = hn;
      h16[base + cc] = (_Float16)hn;
    }
  }
}

// ---- decoder: out = (relu(h@W1+b1)@W2+b2) * M (M=16) ------------------------
__global__ void __launch_bounds__(256) k_decoder(
    const _Float16* __restrict__ h16, const float* __restrict__ x,
    const _Float16* __restrict__ W1T, const float* __restrict__ b1,
    const float* __restrict__ W2, const float* __restrict__ b2,
    float* __restrict__ out) {
  __shared__ __attribute__((aligned(16))) _Float16 sA[16 * W128];
  __shared__ __attribute__((aligned(16))) _Float16 sH[16 * W128];
  int tid = threadIdx.x, lane = tid & 31, wave = tid >> 5;
  int tile = blockIdx.x * 16;
  {
    int row = tid >> 4, ch = tid & 15;
    cp16_g2l(&sA[row * W128 + ch * 8],
             &h16[(size_t)(tile + row) * W128 + ch * 8]);
  }
  async_wait();
  __syncthreads();
  v8f c = wmma_rowtile<128>(sA, W128, W1T, lane, wave);
  store_c_relu_f16(c, sH, b1, lane, wave);
  __syncthreads();
  if (tid < 48) {
    int row = tid / 3, o = tid - row * 3;
    float acc = b2[o];
    for (int k = 0; k < 128; ++k) acc += (float)sH[row * W128 + k] * W2[k * 3 + o];
    float z0 = x[(size_t)(tile + row) * 3 + 1];
    float t0 = x[(size_t)(tile + row) * 3 + 2];
    float M = (z0 <= t0 + DT_C) ? 1.f : 0.f;
    out[(size_t)(tile + row) * 3 + o] = acc * M;
  }
}

// ---- host -------------------------------------------------------------------
extern "C" void kernel_launch(void* const* d_in, const int* in_sizes, int n_in,
                              void* d_out, int out_size, void* d_ws, size_t ws_size,
                              hipStream_t stream) {
  const float* x      = (const float*)d_in[0];
  const float* eattr  = (const float*)d_in[1];
  const int*   eidx   = (const int*)d_in[2];
  const float* enW1 = (const float*)d_in[3],  *enb1 = (const float*)d_in[4];
  const float* enW2 = (const float*)d_in[5],  *enb2 = (const float*)d_in[6];
  const float* eng  = (const float*)d_in[7],  *enbe = (const float*)d_in[8];
  const float* eeW1 = (const float*)d_in[9],  *eeb1 = (const float*)d_in[10];
  const float* eeW2 = (const float*)d_in[11], *eeb2 = (const float*)d_in[12];
  const float* eeg  = (const float*)d_in[13], *eebe = (const float*)d_in[14];
  const float* peW1 = (const float*)d_in[15], *peb1 = (const float*)d_in[16];
  const float* peW2 = (const float*)d_in[17], *peb2 = (const float*)d_in[18];
  const float* peg  = (const float*)d_in[19], *pebe = (const float*)d_in[20];
  const float* pnW1 = (const float*)d_in[21], *pnb1 = (const float*)d_in[22];
  const float* pnW2 = (const float*)d_in[23], *pnb2 = (const float*)d_in[24];
  const float* png  = (const float*)d_in[25], *pnbe = (const float*)d_in[26];
  const float* deW1 = (const float*)d_in[27], *deb1 = (const float*)d_in[28];
  const float* deW2 = (const float*)d_in[29], *deb2 = (const float*)d_in[30];

  char* ws = (char*)d_ws;
  size_t off = 0;
  auto alloc = [&](size_t bytes) -> char* {
    char* p = ws + off;
    off = (off + bytes + 255) & ~(size_t)255;
    return p;
  };
  float*    h     = (float*)alloc((size_t)NN * 128 * 4);
  _Float16* h16   = (_Float16*)alloc((size_t)NN * 128 * 2);
  _Float16* e16   = (_Float16*)alloc((size_t)EE * 128 * 2);
  float*    agg   = (float*)alloc((size_t)NN * 128 * 4);
  _Float16* enW2T = (_Float16*)alloc(16384 * 2);
  _Float16* eeW2T = (_Float16*)alloc(16384 * 2);
  _Float16* dW1T  = (_Float16*)alloc(16384 * 2);
  _Float16* peW1T = (_Float16*)alloc((size_t)4 * 384 * 128 * 2);
  _Float16* peW2T = (_Float16*)alloc((size_t)4 * 16384 * 2);
  _Float16* pnW1T = (_Float16*)alloc((size_t)4 * 256 * 128 * 2);
  _Float16* pnW2T = (_Float16*)alloc((size_t)4 * 16384 * 2);

  auto T = [&](const float* w, _Float16* wt, int K, int Nc) {
    int tot = K * Nc;
    k_transpose_w<<<(tot + 255) / 256, 256, 0, stream>>>(w, wt, K, Nc);
  };
  T(enW2, enW2T, 128, 128);
  T(eeW2, eeW2T, 128, 128);
  T(deW1, dW1T, 128, 128);
  for (int l = 0; l < 4; ++l) {
    T(peW1 + (size_t)l * 384 * 128, peW1T + (size_t)l * 128 * 384, 384, 128);
    T(peW2 + (size_t)l * 16384,     peW2T + (size_t)l * 16384,     128, 128);
    T(pnW1 + (size_t)l * 256 * 128, pnW1T + (size_t)l * 128 * 256, 256, 128);
    T(pnW2 + (size_t)l * 16384,     pnW2T + (size_t)l * 16384,     128, 128);
  }

  k_node_encoder<<<NN / 16, 256, 0, stream>>>(x, enW1, enb1, enW2T, enb2, eng, enbe, h, h16);
  k_edge_encoder<<<EE / 16, 256, 0, stream>>>(eattr, eeW1, eeb1, eeW2T, eeb2, eeg, eebe, e16);

  for (int l = 0; l < 4; ++l) {
    (void)hipMemsetAsync(agg, 0, (size_t)NN * 128 * 4, stream);
    k_edge_layer<<<EE / 32, 256, 0, stream>>>(
        eidx, e16, h16, agg,
        peW1T + (size_t)l * 128 * 384, peb1 + l * 128,
        peW2T + (size_t)l * 16384,     peb2 + l * 128,
        peg + l * 128, pebe + l * 128);
    k_node_layer<<<NN / 32, 256, 0, stream>>>(
        h, h16, agg,
        pnW1T + (size_t)l * 128 * 256, pnb1 + l * 128,
        pnW2T + (size_t)l * 16384,     pnb2 + l * 128,
        png + l * 128, pnbe + l * 128);
  }
  k_decoder<<<NN / 16, 256, 0, stream>>>(h16, x, dW1T, deb1, deW2, deb2, (float*)d_out);
}
